// EnhancedFinancialGAT_64811056496735
// MI455X (gfx1250) — matrix operand order repaired and checked by reading, not compiled, not run on hardware
//
#include <hip/hip_runtime.h>
#include <hip/hip_bf16.h>

typedef __attribute__((ext_vector_type(16))) _Float16 v16h;
typedef __attribute__((ext_vector_type(8)))  float    v8f;

#define LD 324   // padded LDS row stride (floats): 324 % 64 = 4 -> 16 rows hit distinct banks

// Wave-cooperative GEMM for one 16-row M-tile:
//   Out[16, N] = act( In[16, K] @ W[N, K]^T + bias )
// In : row-major, leading dim ldIn (generic pointer: LDS or global), f32
// W  : global row-major [N, K], f32 (converted to f16 in-register)
// Out: LDS row-major, leading dim LD
// Uses v_wmma_f32_16x16x32_f16; fragment layouts per CDNA5 ISA 7.12.2.
__device__ __forceinline__ void gemm16_relu(
    const float* In, int ldIn,
    const float* __restrict__ W,
    const float* __restrict__ bias,
    float* Out,
    int N, int K,
    int wave, int nwaves, int lane, bool do_relu)
{
  const int row = lane & 15;       // M for A-frag, N-in-tile for B/D frags
  const int hi  = lane >> 4;       // lane half
  const int kbA = hi * 8;          // A fragment: K base 0 or 8
  const int kbB = hi * 16;         // unused: B fragment also uses kbA (K base 0 or 8)

  for (int nt = wave; nt < (N >> 4); nt += nwaves) {
    const int col = (nt << 4) + row;             // global output column
    v8f acc = {};
    for (int k0 = 0; k0 < K; k0 += 32) {
      v16h a, b;
      // A 16x32 f16 frag: elems 0..7 -> K = kbA..kbA+7 ; elems 8..15 -> K = 16+kbA..
      const float* arow = In + row * ldIn + k0 + kbA;
      #pragma unroll
      for (int e = 0; e < 8; ++e) {
        a[e]     = (_Float16)arow[e];
        a[e + 8] = (_Float16)arow[e + 16];
      }
      // B 32x16 f16 frag: B[k][n] = W[n][k] -> contiguous K read along W row
      const float* wrow = W + col * K + k0 + kbA;
      #pragma unroll
      for (int e = 0; e < 8; ++e) {
        b[e]     = (_Float16)wrow[e];
        b[e + 8] = (_Float16)wrow[e + 16];
      }
      acc = __builtin_amdgcn_wmma_f32_16x16x32_f16(false, a, false, b,
                                                   (short)0, acc, false, false);
    }
    const float bv = bias[col];
    #pragma unroll
    for (int r = 0; r < 8; ++r) {               // D frag: VGPR r -> M = r (+8 for hi half)
      float v = acc[r] + bv;
      if (do_relu) v = fmaxf(v, 0.0f);
      Out[(hi * 8 + r) * LD + (nt << 4) + row] = v;
    }
  }
}

__global__ __launch_bounds__(256) void fin_gat_collapsed_kernel(
    const float* __restrict__ x,       // [64,64]
    const int*   __restrict__ cidx,    // [64]
    const float* __restrict__ W_in,    // [256,64]
    const float* __restrict__ b_in,    // [256]
    const float* __restrict__ gat_W,   // [3,256,256]
    const float* __restrict__ gat_b,   // [3,256]
    const float* __restrict__ emb,     // [2000,64]
    const float* __restrict__ W_fuse, const float* __restrict__ b_fuse, // [256,320],[256]
    const float* __restrict__ W_p1, const float* __restrict__ b_p1,     // [128,256]
    const float* __restrict__ W_p2, const float* __restrict__ b_p2,     // [64,128]
    const float* __restrict__ W_p3, const float* __restrict__ b_p3,     // [1,64]
    const float* __restrict__ W_d1, const float* __restrict__ b_d1,     // [128,256]
    const float* __restrict__ W_d2, const float* __restrict__ b_d2,     // [64,128]
    const float* __restrict__ W_d3, const float* __restrict__ b_d3,     // [1,64]
    float* __restrict__ out)           // [128] = price[64] ++ direction[64]
{
  __shared__ float bufA[16 * LD];
  __shared__ float bufB[16 * LD];
  const int tid  = threadIdx.x;
  const int lane = tid & 31;
  const int wave = tid >> 5;
  const int nw   = blockDim.x >> 5;          // 8 waves
  const int rowBase = blockIdx.x * 16;       // this block's 16 batch rows

  // h0 = relu(x @ W_in^T + b_in) -> bufA[:, 0:256]
  gemm16_relu(x + rowBase * 64, 64, W_in, b_in, bufA, 256, 64, wave, nw, lane, true);
  __syncthreads();

  // Three collapsed GAT layers: h <- relu(gat_W[l] @ h + gat_b[l])
  gemm16_relu(bufA, LD, gat_W + 0 * 256 * 256, gat_b + 0 * 256, bufB, 256, 256, wave, nw, lane, true);
  __syncthreads();
  gemm16_relu(bufB, LD, gat_W + 1 * 256 * 256, gat_b + 1 * 256, bufA, 256, 256, wave, nw, lane, true);
  __syncthreads();
  gemm16_relu(bufA, LD, gat_W + 2 * 256 * 256, gat_b + 2 * 256, bufB, 256, 256, wave, nw, lane, true);
  __syncthreads();

  // Append company embeddings: bufB[:, 256:320] = emb[cidx[row]]
  for (int i = tid; i < 16 * 64; i += blockDim.x) {
    const int r = i >> 6, c = i & 63;
    bufB[r * LD + 256 + c] = emb[cidx[rowBase + r] * 64 + c];
  }
  __syncthreads();

  // fused = relu([h | emb] @ W_fuse^T + b_fuse) -> bufA[:, 0:256]
  gemm16_relu(bufB, LD, W_fuse, b_fuse, bufA, 256, 320, wave, nw, lane, true);
  __syncthreads();

  // price branch: p1 -> bufB[:,0:128], p2 -> bufB[:,128:192]
  gemm16_relu(bufA, LD, W_p1, b_p1, bufB, 128, 256, wave, nw, lane, true);
  __syncthreads();
  gemm16_relu(bufB, LD, W_p2, b_p2, bufB + 128, 64, 128, wave, nw, lane, true);
  __syncthreads();

  // direction branch: d1 -> bufB[:,192:320], d2 -> bufB[:,0:64] (p1 dead now)
  gemm16_relu(bufA, LD, W_d1, b_d1, bufB + 192, 128, 256, wave, nw, lane, true);
  __syncthreads();
  gemm16_relu(bufB + 192, LD, W_d2, b_d2, bufB, 64, 128, wave, nw, lane, true);
  __syncthreads();

  // Heads (N=1: scalar dot per row)
  if (tid < 16) {
    const int r = tid;
    float s = b_p3[0];
    #pragma unroll 8
    for (int k = 0; k < 64; ++k) s += bufB[r * LD + 128 + k] * W_p3[k];
    out[rowBase + r] = s;
  } else if (tid < 32) {
    const int r = tid - 16;
    float s = b_d3[0];
    #pragma unroll 8
    for (int k = 0; k < 64; ++k) s += bufB[r * LD + k] * W_d3[k];
    out[64 + rowBase + r] = 1.0f / (1.0f + __expf(-s));
  }
}

extern "C" void kernel_launch(void* const* d_in, const int* in_sizes, int n_in,
                              void* d_out, int out_size, void* d_ws, size_t ws_size,
                              hipStream_t stream) {
  (void)in_sizes; (void)n_in; (void)out_size; (void)d_ws; (void)ws_size;
  const float* x      = (const float*)d_in[0];
  const int*   cidx   = (const int*)  d_in[1];
  // d_in[2] edge_index, d_in[3] edge_attr: cancel out of the collapsed math
  const float* W_in   = (const float*)d_in[4];
  const float* b_in   = (const float*)d_in[5];
  const float* gat_W  = (const float*)d_in[6];
  // d_in[7..10] att_src/att_dst/We/att_edge: attention weights sum to 1 -> unused
  const float* gat_b  = (const float*)d_in[11];
  const float* emb    = (const float*)d_in[12];
  const float* W_fuse = (const float*)d_in[13];
  const float* b_fuse = (const float*)d_in[14];
  const float* W_p1   = (const float*)d_in[15];
  const float* b_p1   = (const float*)d_in[16];
  const float* W_p2   = (const float*)d_in[17];
  const float* b_p2   = (const float*)d_in[18];
  const float* W_p3   = (const float*)d_in[19];
  const float* b_p3   = (const float*)d_in[20];
  const float* W_d1   = (const float*)d_in[21];
  const float* b_d1   = (const float*)d_in[22];
  const float* W_d2   = (const float*)d_in[23];
  const float* b_d2   = (const float*)d_in[24];
  const float* W_d3   = (const float*)d_in[25];
  const float* b_d3   = (const float*)d_in[26];

  fin_gat_collapsed_kernel<<<dim3(4), dim3(256), 0, stream>>>(
      x, cidx, W_in, b_in, gat_W, gat_b, emb, W_fuse, b_fuse,
      W_p1, b_p1, W_p2, b_p2, W_p3, b_p3,
      W_d1, b_d1, W_d2, b_d2, W_d3, b_d3,
      (float*)d_out);
}
